// MultiHeadAttention_34256659153517
// MI455X (gfx1250) — matrix operand order repaired and checked by reading, not compiled, not run on hardware
//
#include <hip/hip_runtime.h>

#define DEVINL __device__ __forceinline__

typedef __attribute__((ext_vector_type(16))) __bf16 bfx16;
typedef __attribute__((ext_vector_type(8)))  __bf16 bfx8;
typedef __attribute__((ext_vector_type(8)))  float  fx8;

namespace {
constexpr int BATCH = 4;
constexpr int SEQL  = 1024;
constexpr int DMODEL = 1024;
constexpr int NHEAD = 16;
constexpr int DHEAD = 64;
constexpr int VOCAB = 129;          // 2*64+1 relative buckets
constexpr int VPN   = 144;          // vocab padded to N-tile multiple (9*16)
constexpr int VPK   = 160;          // vocab padded to K-tile multiple (5*32)
constexpr int ROWS  = BATCH * SEQL; // 4096
constexpr int BH    = BATCH * NHEAD; // 64
}

// ---------- bf16 storage helpers (no reliance on __bf16 arithmetic) ----------
DEVINL __bf16 f2bf(float f) {
  union { float f; unsigned u; } v; v.f = f;
  unsigned r = v.u + 0x7FFFu + ((v.u >> 16) & 1u);   // round-to-nearest-even
  unsigned short h = (unsigned short)(r >> 16);
  __bf16 out; __builtin_memcpy(&out, &h, 2);
  return out;
}
DEVINL float bf2f(__bf16 b) {
  unsigned short h; __builtin_memcpy(&h, &b, 2);
  union { unsigned u; float f; } v; v.u = ((unsigned)h) << 16;
  return v.f;
}

// ---------- WMMA fragment loaders (contiguous-K order shared by A and B, wave32) ----------
// A 16x32 bf16: lane l -> row (l&15); lanes 0-15 hold K=0..15; lanes 16-31 hold K=16..31
DEVINL bfx16 load_a16x32(const __bf16* base, int ld) {
  const int lane = threadIdx.x & 31;
  const __bf16* p = base + (lane & 15) * ld + (lane >> 4) * 16;
  union { bfx16 v; bfx8 h[2]; } u;
  u.h[0] = *(const bfx8*)(p);
  u.h[1] = *(const bfx8*)(p + 8);
  return u.v;
}
// B 32x16 bf16 from Bt stored row-major [N, K]: lane l -> col (l&15), K base (l>>4)*16, 16 contiguous
DEVINL bfx16 load_bt32x16(const __bf16* bt, int ld) {
  const int lane = threadIdx.x & 31;
  return *(const bfx16*)(bt + (lane & 15) * ld + (lane >> 4) * 16);
}
DEVINL fx8 wmma_bf16(bfx16 a, bfx16 b, fx8 c) {
  return __builtin_amdgcn_wmma_f32_16x16x32_bf16(false, a, false, b, (short)0, c,
                                                 false, false);
}
DEVINL fx8 fx8_zero() {
  fx8 c;
#pragma unroll
  for (int i = 0; i < 8; ++i) c[i] = 0.f;
  return c;
}

// ---------------------- prep kernels ----------------------
__global__ void cast_bf16_kernel(const float* __restrict__ src,
                                 __bf16* __restrict__ dst, int n) {
  for (int i = blockIdx.x * blockDim.x + threadIdx.x; i < n;
       i += gridDim.x * blockDim.x)
    dst[i] = f2bf(src[i]);
}

// W [1024,1024] fp32 row-major -> Wt [1024,1024] bf16 (Wt[n][k] = W[k][n])
__global__ void transpose_cast_kernel(const float* __restrict__ w,
                                      __bf16* __restrict__ wt) {
  int idx = blockIdx.x * blockDim.x + threadIdx.x;   // 1M threads
  int r = idx >> 10, c = idx & 1023;
  wt[c * DMODEL + r] = f2bf(w[idx]);
}

// rel_k_table [129,64] -> bf16 [144,64], zero padded rows (acts as Bt for Q@rel_k^T)
__global__ void build_relk_kernel(const float* __restrict__ t,
                                  __bf16* __restrict__ o) {
  int idx = blockIdx.x * blockDim.x + threadIdx.x;
  if (idx >= VPN * DHEAD) return;
  int r = idx / DHEAD, d = idx % DHEAD;
  o[idx] = f2bf(r < VOCAB ? t[r * DHEAD + d] : 0.f);
}
// rel_v_table [129,64] -> transposed bf16 [64,160], zero padded K (Bt for Pr@rel_v)
__global__ void build_relv_kernel(const float* __restrict__ t,
                                  __bf16* __restrict__ o) {
  int idx = blockIdx.x * blockDim.x + threadIdx.x;
  if (idx >= DHEAD * VPK) return;
  int d = idx / VPK, r = idx % VPK;
  o[idx] = f2bf(r < VOCAB ? t[r * DHEAD + d] : 0.f);
}

// ---------------------- fused QKV projection ----------------------
// out = x @ W + b for W in {wq,wk,wv}; q,k stored head-major [BH, L, 64] bf16,
// v stored transposed [BH, 64, L] bf16 (B-operand layout for P@V).
// Register-blocked: one wave computes a 32(M) x 64(N) strip (8 WMMA tiles),
// reusing 2 A-frags x 4 B-frags per K-step -> 1.5 b128 loads per WMMA.
__global__ __launch_bounds__(256) void qkv_gemm_kernel(
    const __bf16* __restrict__ xb,
    const __bf16* __restrict__ wqt, const __bf16* __restrict__ wkt,
    const __bf16* __restrict__ wvt,
    const float* __restrict__ bq, const float* __restrict__ bk,
    const float* __restrict__ bv,
    __bf16* __restrict__ qo, __bf16* __restrict__ ko,
    __bf16* __restrict__ vto) {
  const int proj = blockIdx.y;
  const __bf16* wt  = proj == 0 ? wqt : proj == 1 ? wkt : wvt;
  const float*  bias = proj == 0 ? bq : proj == 1 ? bk : bv;
  const int wave = threadIdx.x >> 5, lane = threadIdx.x & 31;
  const int job = blockIdx.x * 8 + wave;        // 2048 wave-jobs
  const int mt = job >> 4;                      // 128 M-tiles of 32 rows
  const int nt = job & 15;                      // 16 N-strips of 64 cols
  const int nn = lane & 15, mh = (lane >> 4) * 8;

  fx8 c[2][4];
  #pragma unroll
  for (int j = 0; j < 2; ++j)
    #pragma unroll
    for (int bi = 0; bi < 4; ++bi) c[j][bi] = fx8_zero();

  const __bf16* abase = xb + mt * 32 * DMODEL;
  const __bf16* bbase = wt + nt * 64 * DMODEL;
  for (int k0 = 0; k0 < DMODEL; k0 += 32) {
    if (k0 + 64 < DMODEL) {   // gfx1250 global_prefetch_b8, two K-tiles ahead
      __builtin_prefetch(abase + (lane & 15) * DMODEL + k0 + 64, 0, 1);
      __builtin_prefetch(bbase + ((lane & 15) + 16) * DMODEL + k0 + 64, 0, 1);
    }
    bfx16 a0 = load_a16x32(abase + k0, DMODEL);
    bfx16 a1 = load_a16x32(abase + 16 * DMODEL + k0, DMODEL);
    #pragma unroll
    for (int bi = 0; bi < 4; ++bi) {
      bfx16 b = load_bt32x16(bbase + bi * 16 * DMODEL + k0, DMODEL);
      c[0][bi] = wmma_bf16(a0, b, c[0][bi]);
      c[1][bi] = wmma_bf16(a1, b, c[1][bi]);
    }
  }
  #pragma unroll
  for (int bi = 0; bi < 4; ++bi) {
    const int col = nt * 64 + bi * 16 + nn;
    const float bia = bias[col];
    const int h = col >> 6, d = col & 63;
    #pragma unroll
    for (int j = 0; j < 2; ++j) {
      #pragma unroll
      for (int i = 0; i < 8; ++i) {
        int m = mt * 32 + j * 16 + mh + i;
        int b_ = m >> 10, l = m & 1023;
        int bh = b_ * NHEAD + h;
        __bf16 v = f2bf(c[j][bi][i] + bia);
        if (proj == 0)      qo[(bh * SEQL + l) * DHEAD + d] = v;
        else if (proj == 1) ko[(bh * SEQL + l) * DHEAD + d] = v;
        else                vto[(bh * DHEAD + d) * SEQL + l] = v;
      }
    }
  }
}

// ---------------------- relative score GEMM: srel = Q @ rel_k^T ----------------------
// per (b,h): [1024, 64] @ [64, 144pad] -> fp32 srel [BH, L, 144]
__global__ __launch_bounds__(256) void relscore_kernel(
    const __bf16* __restrict__ qb, const __bf16* __restrict__ relk,
    float* __restrict__ srel) {
  const int wave = threadIdx.x >> 5, lane = threadIdx.x & 31;
  const int tid8 = blockIdx.x * 8 + wave;       // 64*64*9 = 36864 tiles
  const int bh = tid8 / 576;
  const int rem = tid8 % 576;
  const int mt = rem / 9, nt = rem % 9;
  const int nn = lane & 15, mh = (lane >> 4) * 8;

  fx8 c = fx8_zero();
  const __bf16* abase = qb + (bh * SEQL + mt * 16) * DHEAD;
  const __bf16* bbase = relk + nt * 16 * DHEAD;
  #pragma unroll
  for (int k0 = 0; k0 < DHEAD; k0 += 32) {
    bfx16 a = load_a16x32(abase + k0, DHEAD);
    bfx16 b = load_bt32x16(bbase + k0, DHEAD);
    c = wmma_bf16(a, b, c);
  }
  #pragma unroll
  for (int i = 0; i < 8; ++i) {
    int m = mt * 16 + mh + i;
    srel[(bh * SEQL + m) * VPN + nt * 16 + nn] = c[i];
  }
}

// ---------------------- fused attention ----------------------
// One block = one (b,h,16-row q block). Scores kept as a 16x1024 bf16 strip in LDS.
__global__ __launch_bounds__(256) void attn_kernel(
    const __bf16* __restrict__ qb, const __bf16* __restrict__ kb,
    const __bf16* __restrict__ vtb, const float* __restrict__ srel,
    const float* __restrict__ mask, const __bf16* __restrict__ relvt,
    __bf16* __restrict__ ctxb) {
  __shared__ __bf16 Sp[16][SEQL];        // 32 KB score/prob strip
  __shared__ float redmax[16][8];
  __shared__ float rowmax[16];
  __shared__ float rowsum[16];
  __shared__ float ctxacc[16][64];
  __shared__ __bf16 Pr[16][VPK];         // aggregated rel-bucket probs

  const int bh = blockIdx.x >> 6;        // gridDim.x = 64*64
  const int qt = blockIdx.x & 63;
  const int b_ = bh >> 4, h = bh & 15;
  const int q0 = qt * 16;
  const int tid = threadIdx.x;
  const int wave = tid >> 5, lane = tid & 31;
  const int nn = lane & 15, mh = (lane >> 4) * 8;

  if (tid < 16) rowsum[tid] = 0.f;
  for (int i = tid; i < 16 * 64; i += 256) ctxacc[i >> 6][i & 63] = 0.f;

  // ---- pass 1: S = Q K^T + srel_gather + mask, per-wave 128-key stripe ----
  bfx16 a0 = load_a16x32(qb + (bh * SEQL + q0) * DHEAD, DHEAD);
  bfx16 a1 = load_a16x32(qb + (bh * SEQL + q0) * DHEAD + 32, DHEAD);
  float pm[8];
  #pragma unroll
  for (int i = 0; i < 8; ++i) pm[i] = -3.0e38f;
  for (int t = 0; t < 8; ++t) {
    const int k0 = wave * 128 + t * 16;
    bfx16 b0 = load_bt32x16(kb + (bh * SEQL + k0) * DHEAD, DHEAD);
    bfx16 b1 = load_bt32x16(kb + (bh * SEQL + k0) * DHEAD + 32, DHEAD);
    fx8 c = fx8_zero();
    c = wmma_bf16(a0, b0, c);
    c = wmma_bf16(a1, b1, c);
    const int kk = k0 + nn;
    const float mval = mask[b_ * SEQL + kk] * (-1e9f);
    #pragma unroll
    for (int i = 0; i < 8; ++i) {
      int qa = q0 + mh + i;
      int d = kk - qa; d = d < -64 ? -64 : (d > 64 ? 64 : d);
      float s = c[i] + mval + srel[(bh * SEQL + qa) * VPN + (d + 64)];
      pm[i] = fmaxf(pm[i], s);
      Sp[mh + i][kk] = f2bf(s);
    }
  }
  #pragma unroll
  for (int msk = 1; msk < 16; msk <<= 1) {
    #pragma unroll
    for (int i = 0; i < 8; ++i) pm[i] = fmaxf(pm[i], __shfl_xor(pm[i], msk, 32));
  }
  if (nn == 0) {
    #pragma unroll
    for (int i = 0; i < 8; ++i) redmax[mh + i][wave] = pm[i];
  }
  __syncthreads();
  if (tid < 16) {
    float mx = redmax[tid][0];
    #pragma unroll
    for (int w = 1; w < 8; ++w) mx = fmaxf(mx, redmax[tid][w]);
    rowmax[tid] = mx;
  }
  __syncthreads();

  // ---- pass 2: exp + row sums (unnormalized probs kept in Sp) ----
  {
    const int m = tid >> 4, seg = tid & 15;
    const float lm = rowmax[m];
    float ls = 0.f;
    for (int k = seg * 64; k < seg * 64 + 64; ++k) {
      float p = __expf(bf2f(Sp[m][k]) - lm);
      Sp[m][k] = f2bf(p);
      ls += p;
    }
    atomicAdd(&rowsum[m], ls);
  }
  __syncthreads();

  // ---- pass 3: ctx += P @ V, per-wave K stripe, LDS reduce ----
  for (int nt = 0; nt < 4; ++nt) {
    fx8 c = fx8_zero();
    for (int ks = 0; ks < 4; ++ks) {
      const int kofs = wave * 128 + ks * 32;
      bfx16 a = load_a16x32(&Sp[0][0] + kofs, SEQL);
      bfx16 b = load_bt32x16(vtb + (bh * DHEAD + nt * 16) * SEQL + kofs, SEQL);
      c = wmma_bf16(a, b, c);
    }
    #pragma unroll
    for (int i = 0; i < 8; ++i) atomicAdd(&ctxacc[mh + i][nt * 16 + nn], c[i]);
  }
  __syncthreads();

  // ---- pass 4: aggregate probs per relative bucket ----
  {
    const int m = tid >> 4, j = tid & 15;
    const int qa = q0 + m;
    for (int r = j; r < VPK; r += 16) {
      float v = 0.f;
      if (r == 0) {
        int ke = qa - 64;
        for (int k = 0; k <= ke; ++k) v += bf2f(Sp[m][k]);
      } else if (r < 128) {
        int k = qa + r - 64;
        if (k >= 0 && k < SEQL) v = bf2f(Sp[m][k]);
      } else if (r == 128) {
        for (int k = qa + 64; k < SEQL; ++k) v += bf2f(Sp[m][k]);
      }
      Pr[m][r] = f2bf(v);
    }
  }
  __syncthreads();

  // ---- pass 5: ctx += Pr @ rel_v (waves 0..3, one 16-col tile each) ----
  if (wave < 4) {
    fx8 c = fx8_zero();
    #pragma unroll
    for (int ks = 0; ks < 5; ++ks) {
      bfx16 a = load_a16x32(&Pr[0][0] + ks * 32, VPK);
      bfx16 b = load_bt32x16(relvt + (wave * 16) * VPK + ks * 32, VPK);
      c = wmma_bf16(a, b, c);
    }
    #pragma unroll
    for (int i = 0; i < 8; ++i) ctxacc[mh + i][wave * 16 + nn] += c[i];
  }
  __syncthreads();

  // ---- pass 6: normalize + store combined-head ctx (bf16) ----
  {
    const int m = tid >> 4, j = tid & 15;
    const float inv = 1.f / rowsum[m];
    const int row = b_ * SEQL + q0 + m;
    #pragma unroll
    for (int d = j * 4; d < j * 4 + 4; ++d)
      ctxb[row * DMODEL + h * DHEAD + d] = f2bf(ctxacc[m][d] * inv);
  }
}

// ---------------------- output projection ----------------------
// Register-blocked 32x64 per wave, same scheme as qkv_gemm.
__global__ __launch_bounds__(256) void outproj_kernel(
    const __bf16* __restrict__ ctxb, const __bf16* __restrict__ wot,
    const float* __restrict__ bo, float* __restrict__ out) {
  const int wave = threadIdx.x >> 5, lane = threadIdx.x & 31;
  const int job = blockIdx.x * 8 + wave;        // 2048 wave-jobs
  const int mt = job >> 4, nt = job & 15;
  const int nn = lane & 15, mh = (lane >> 4) * 8;

  fx8 c[2][4];
  #pragma unroll
  for (int j = 0; j < 2; ++j)
    #pragma unroll
    for (int bi = 0; bi < 4; ++bi) c[j][bi] = fx8_zero();

  const __bf16* abase = ctxb + mt * 32 * DMODEL;
  const __bf16* bbase = wot + nt * 64 * DMODEL;
  for (int k0 = 0; k0 < DMODEL; k0 += 32) {
    if (k0 + 64 < DMODEL) {
      __builtin_prefetch(abase + (lane & 15) * DMODEL + k0 + 64, 0, 1);
      __builtin_prefetch(bbase + ((lane & 15) + 16) * DMODEL + k0 + 64, 0, 1);
    }
    bfx16 a0 = load_a16x32(abase + k0, DMODEL);
    bfx16 a1 = load_a16x32(abase + 16 * DMODEL + k0, DMODEL);
    #pragma unroll
    for (int bi = 0; bi < 4; ++bi) {
      bfx16 b = load_bt32x16(bbase + bi * 16 * DMODEL + k0, DMODEL);
      c[0][bi] = wmma_bf16(a0, b, c[0][bi]);
      c[1][bi] = wmma_bf16(a1, b, c[1][bi]);
    }
  }
  #pragma unroll
  for (int bi = 0; bi < 4; ++bi) {
    const float bia = bo[nt * 64 + bi * 16 + nn];
    #pragma unroll
    for (int j = 0; j < 2; ++j) {
      #pragma unroll
      for (int i = 0; i < 8; ++i)
        out[(mt * 32 + j * 16 + mh + i) * DMODEL + nt * 64 + bi * 16 + nn] =
            c[j][bi][i] + bia;
    }
  }
}

// ---------------------- launcher ----------------------
extern "C" void kernel_launch(void* const* d_in, const int* in_sizes, int n_in,
                              void* d_out, int out_size, void* d_ws, size_t ws_size,
                              hipStream_t stream) {
  const float* x    = (const float*)d_in[0];
  const float* mask = (const float*)d_in[1];
  const float* wq   = (const float*)d_in[2];
  const float* bq   = (const float*)d_in[3];
  const float* wk   = (const float*)d_in[4];
  const float* bk   = (const float*)d_in[5];
  const float* wv   = (const float*)d_in[6];
  const float* bv   = (const float*)d_in[7];
  const float* wo   = (const float*)d_in[8];
  const float* bo   = (const float*)d_in[9];
  const float* relk = (const float*)d_in[10];
  const float* relv = (const float*)d_in[11];

  char* ws = (char*)d_ws;
  size_t off = 0;
  auto alloc = [&](size_t bytes) -> void* {
    void* p = ws + off;
    off += (bytes + 255) & ~(size_t)255;
    return p;
  };
  __bf16* xb    = (__bf16*)alloc((size_t)ROWS * DMODEL * 2);
  __bf16* wqt   = (__bf16*)alloc((size_t)DMODEL * DMODEL * 2);
  __bf16* wkt   = (__bf16*)alloc((size_t)DMODEL * DMODEL * 2);
  __bf16* wvt   = (__bf16*)alloc((size_t)DMODEL * DMODEL * 2);
  __bf16* wot   = (__bf16*)alloc((size_t)DMODEL * DMODEL * 2);
  __bf16* relkb = (__bf16*)alloc((size_t)VPN * DHEAD * 2);
  __bf16* relvt = (__bf16*)alloc((size_t)DHEAD * VPK * 2);
  __bf16* qb    = (__bf16*)alloc((size_t)BH * SEQL * DHEAD * 2);
  __bf16* kb    = (__bf16*)alloc((size_t)BH * SEQL * DHEAD * 2);
  __bf16* vtb   = (__bf16*)alloc((size_t)BH * DHEAD * SEQL * 2);
  float*  srel  = (float*)alloc((size_t)BH * SEQL * VPN * 4);
  __bf16* ctxb  = (__bf16*)alloc((size_t)ROWS * DMODEL * 2);

  cast_bf16_kernel<<<4096, 256, 0, stream>>>(x, xb, ROWS * DMODEL);
  transpose_cast_kernel<<<4096, 256, 0, stream>>>(wq, wqt);
  transpose_cast_kernel<<<4096, 256, 0, stream>>>(wk, wkt);
  transpose_cast_kernel<<<4096, 256, 0, stream>>>(wv, wvt);
  transpose_cast_kernel<<<4096, 256, 0, stream>>>(wo, wot);
  build_relk_kernel<<<(VPN * DHEAD + 255) / 256, 256, 0, stream>>>(relk, relkb);
  build_relv_kernel<<<(DHEAD * VPK + 255) / 256, 256, 0, stream>>>(relv, relvt);

  qkv_gemm_kernel<<<dim3(256, 3), 256, 0, stream>>>(
      xb, wqt, wkt, wvt, bq, bk, bv, qb, kb, vtb);
  relscore_kernel<<<4608, 256, 0, stream>>>(qb, relkb, srel);
  attn_kernel<<<BH * (SEQL / 16), 256, 0, stream>>>(
      qb, kb, vtb, srel, mask, relvt, ctxb);
  outproj_kernel<<<256, 256, 0, stream>>>(ctxb, wot, bo, (float*)d_out);
}
